// MultiLevelContrastiveLoss_52080773431814
// MI455X (gfx1250) — matrix operand order repaired and checked by reading, not compiled, not run on hardware
//
#include <hip/hip_runtime.h>
#include <hip/hip_fp16.h>

typedef __attribute__((ext_vector_type(16))) _Float16 v16h;
typedef __attribute__((ext_vector_type(8)))  float    v8f;

#define B_SZ   4096
#define D_SZ   256
#define TWO_B  8192
#define INV_T  20.0f
#define NRPAIR 128            // B / 32 : row-tile PAIRS (32 rows per wave)
#define NCTILE 512            // 2B/16 column tiles
#define NCHUNK 32
#define TILES_PER_CHUNK 16    // NCTILE / NCHUNK
#define KEXP   28.853900817779268f   // 20 / ln(2): exp(20c-20) = 2^(K*c - K)

// ---------------------------------------------------------------------------
// Kernel 1: normalize rows (f32), convert to f16, scatter into WMMA
// fragment layouts (contiguous k = 16*hi + elem). One wave per row; lane L handles k=L*8..+7.
// ---------------------------------------------------------------------------
__global__ void __launch_bounds__(256)
prep_norm_kernel(const float* __restrict__ z1, const float* __restrict__ z2,
                 _Float16* __restrict__ aFrag, _Float16* __restrict__ bFrag) {
    int rid  = (int)((blockIdx.x * blockDim.x + threadIdx.x) >> 5); // row 0..8191
    int lane = threadIdx.x & 31;
    if (rid >= TWO_B) return;
    const float* src = (rid < B_SZ) ? (z1 + (size_t)rid * D_SZ)
                                    : (z2 + (size_t)(rid - B_SZ) * D_SZ);
    float v[8];
    float ss = 0.f;
#pragma unroll
    for (int t = 0; t < 8; ++t) { v[t] = src[lane * 8 + t]; ss += v[t] * v[t]; }
#pragma unroll
    for (int off = 16; off > 0; off >>= 1) ss += __shfl_xor(ss, off, 32);
    float scale = 1.0f / fmaxf(sqrtf(ss), 1e-12f);

    int ct = rid >> 4;          // tile index
    int N  = rid & 15;          // position within tile
#pragma unroll
    for (int t = 0; t < 8; ++t) {
        int k  = lane * 8 + t;
        _Float16 h = (_Float16)(v[t] * scale);
        int ks = k >> 5;
        int kk = k & 31;
        { // B-matrix layout (32x16): lane = N + 16*(kk/16), elem = kk%16
            int hi = kk >> 4;
            int e  = kk & 15;
            bFrag[(((size_t)ct * 8 + ks) * 32 + (N + 16 * hi)) * 16 + e] = h;
        }
        if (rid < B_SZ) { // A-matrix layout (16x32)
            int hi = kk >> 4;
            int e  = kk & 15;
            aFrag[(((size_t)ct * 8 + ks) * 32 + (N + 16 * hi)) * 16 + e] = h;
        }
    }
}

// ---------------------------------------------------------------------------
// Kernel 2: per-column level masks as float pairs {m1,m2} for FMA accumulation.
// ---------------------------------------------------------------------------
__global__ void __launch_bounds__(256)
mask_kernel(const int* __restrict__ mt1, const int* __restrict__ mt2,
            float2* __restrict__ maskf) {
    int i = blockIdx.x * blockDim.x + threadIdx.x;
    if (i >= B_SZ) return;
    int a = mt1[i], b = mt2[i];
    float2 m;
    m.x = (a == b) ? 1.f : 0.f;
    m.y = (a == 0 && b == 0) ? 1.f : 0.f;
    maskf[i]        = m;
    maskf[i + B_SZ] = m;
}

// ---------------------------------------------------------------------------
// Kernel 3: fused WMMA GEMM + masked exp-sum, fixed shift 20 (= 1/T) since
// normalized dot <= 1. One wave per (row-tile-pair, column chunk): A for
// 32 rows held in 128 VGPRs; each B fragment feeds TWO wmma ops.
// C layout: lane = N + 16*hi, VGPR r holds row M = hi*8 + r.
// ---------------------------------------------------------------------------
__global__ void __launch_bounds__(32)
gemm_lse_kernel(const _Float16* __restrict__ aFrag, const _Float16* __restrict__ bFrag,
                const float2* __restrict__ maskf,
                float* __restrict__ partial, float* __restrict__ posbuf) {
    int q     = blockIdx.x & (NRPAIR - 1);     // row-tile pair
    int chunk = blockIdx.x >> 7;               // column chunk
    int lane  = threadIdx.x;
    int rt0 = 2 * q, rt1 = 2 * q + 1;

    v16h a0[8], a1[8];
#pragma unroll
    for (int ks = 0; ks < 8; ++ks) {
        a0[ks] = *(const v16h*)(aFrag + (((size_t)rt0 * 8 + ks) * 32 + lane) * 16);
        a1[ks] = *(const v16h*)(aFrag + (((size_t)rt1 * 8 + ks) * 32 + lane) * 16);
    }

    float s0a[8] = {}, s1a[8] = {}, s2a[8] = {};
    float s0b[8] = {}, s1b[8] = {}, s2b[8] = {};
    int n = lane & 15, hi = lane >> 4;

    for (int tt = 0; tt < TILES_PER_CHUNK; ++tt) {
        int ct = chunk * TILES_PER_CHUNK + tt;
        // prefetch next tile's B fragments toward the WGP
        if (tt + 1 < TILES_PER_CHUNK)
            __builtin_prefetch(bFrag + ((size_t)(ct + 1) * 8) * 512 + lane * 16, 0, 3);

        v8f c0 = {}, c1 = {};
#pragma unroll
        for (int ks = 0; ks < 8; ++ks) {
            v16h b = *(const v16h*)(bFrag + (((size_t)ct * 8 + ks) * 32 + lane) * 16);
            c0 = __builtin_amdgcn_wmma_f32_16x16x32_f16(false, a0[ks], false, b,
                                                        (short)0, c0, false, false);
            c1 = __builtin_amdgcn_wmma_f32_16x16x32_f16(false, a1[ks], false, b,
                                                        (short)0, c1, false, false);
        }

        int j = ct * 16 + n;
        float2 m = maskf[j];

        // ---- epilogue for row tile 0 ----
        if (ct == rt0 || ct == rt0 + (B_SZ / 16)) {  // rare: diag or positive tile
            bool isdiag = (ct == rt0);
#pragma unroll
            for (int r = 0; r < 8; ++r) {
                float w = __builtin_amdgcn_exp2f(__builtin_fmaf(c0[r], KEXP, -KEXP));
                if (isdiag && (n == hi * 8 + r)) w = 0.f;
                if (!isdiag && (n == hi * 8 + r))               // j == i + B
                    posbuf[rt0 * 16 + n] = c0[r] * INV_T;
                s0a[r] += w;
                s1a[r] = __builtin_fmaf(m.x, w, s1a[r]);
                s2a[r] = __builtin_fmaf(m.y, w, s2a[r]);
            }
        } else {
#pragma unroll
            for (int r = 0; r < 8; ++r) {
                float w = __builtin_amdgcn_exp2f(__builtin_fmaf(c0[r], KEXP, -KEXP));
                s0a[r] += w;
                s1a[r] = __builtin_fmaf(m.x, w, s1a[r]);
                s2a[r] = __builtin_fmaf(m.y, w, s2a[r]);
            }
        }
        // ---- epilogue for row tile 1 ----
        if (ct == rt1 || ct == rt1 + (B_SZ / 16)) {
            bool isdiag = (ct == rt1);
#pragma unroll
            for (int r = 0; r < 8; ++r) {
                float w = __builtin_amdgcn_exp2f(__builtin_fmaf(c1[r], KEXP, -KEXP));
                if (isdiag && (n == hi * 8 + r)) w = 0.f;
                if (!isdiag && (n == hi * 8 + r))
                    posbuf[rt1 * 16 + n] = c1[r] * INV_T;
                s0b[r] += w;
                s1b[r] = __builtin_fmaf(m.x, w, s1b[r]);
                s2b[r] = __builtin_fmaf(m.y, w, s2b[r]);
            }
        } else {
#pragma unroll
            for (int r = 0; r < 8; ++r) {
                float w = __builtin_amdgcn_exp2f(__builtin_fmaf(c1[r], KEXP, -KEXP));
                s0b[r] += w;
                s1b[r] = __builtin_fmaf(m.x, w, s1b[r]);
                s2b[r] = __builtin_fmaf(m.y, w, s2b[r]);
            }
        }
    }

    // Reduce across the 16 lanes of each half-wave (columns of each row).
#pragma unroll
    for (int r = 0; r < 8; ++r) {
#pragma unroll
        for (int off = 8; off > 0; off >>= 1) {
            s0a[r] += __shfl_xor(s0a[r], off, 16);
            s1a[r] += __shfl_xor(s1a[r], off, 16);
            s2a[r] += __shfl_xor(s2a[r], off, 16);
            s0b[r] += __shfl_xor(s0b[r], off, 16);
            s1b[r] += __shfl_xor(s1b[r], off, 16);
            s2b[r] += __shfl_xor(s2b[r], off, 16);
        }
    }

    if (n == 0) {  // lanes 0 and 16 write rows (hi*8 .. hi*8+7) of each tile
        size_t base = (size_t)chunk * 3 * B_SZ;
#pragma unroll
        for (int r = 0; r < 8; ++r) {
            int i0 = rt0 * 16 + hi * 8 + r;
            int i1 = rt1 * 16 + hi * 8 + r;
            partial[base + 0 * B_SZ + i0] = s0a[r];
            partial[base + 1 * B_SZ + i0] = s1a[r];
            partial[base + 2 * B_SZ + i0] = s2a[r];
            partial[base + 0 * B_SZ + i1] = s0b[r];
            partial[base + 1 * B_SZ + i1] = s1b[r];
            partial[base + 2 * B_SZ + i1] = s2b[r];
        }
    }
}

// ---------------------------------------------------------------------------
// Kernel 4: finalize. Sum chunk partials, add the prepended exp(pos-20),
// lse = 20 + log(sum), per-row loss = lse - pos, masked means, weighted total.
// ---------------------------------------------------------------------------
__global__ void __launch_bounds__(256)
final_kernel(const float* __restrict__ partial, const float* __restrict__ posbuf,
             const float2* __restrict__ maskf, float* __restrict__ out) {
    __shared__ float red[5][256];
    int t = threadIdx.x;
    float l0 = 0.f, l1 = 0.f, l2 = 0.f, c1 = 0.f, c2 = 0.f;
    for (int i = t; i < B_SZ; i += 256) {
        float S0 = 0.f, S1 = 0.f, S2 = 0.f;
        for (int ch = 0; ch < NCHUNK; ++ch) {
            size_t base = (size_t)ch * 3 * B_SZ;
            S0 += partial[base + i];
            S1 += partial[base + B_SZ + i];
            S2 += partial[base + 2 * B_SZ + i];
        }
        float  p  = posbuf[i];
        float  ep = __expf(p - INV_T);      // prepended positive term
        float2 m  = maskf[i];
        l0 += INV_T + logf(S0 + ep) - p;
        l1 += m.x * (INV_T + logf(S1 + ep) - p);  c1 += m.x;
        l2 += m.y * (INV_T + logf(S2 + ep) - p);  c2 += m.y;
    }
    red[0][t] = l0; red[1][t] = l1; red[2][t] = l2; red[3][t] = c1; red[4][t] = c2;
    __syncthreads();
    for (int off = 128; off > 0; off >>= 1) {
        if (t < off)
            for (int qq = 0; qq < 5; ++qq) red[qq][t] += red[qq][t + off];
        __syncthreads();
    }
    if (t == 0) {
        float matter = red[0][0] / (float)B_SZ;
        float cosmo  = red[3][0] > 0.f ? red[1][0] / red[3][0] : 0.f;
        float comp   = red[4][0] > 0.f ? red[2][0] / red[4][0] : 0.f;
        out[0] = 1.0f * matter + 0.7f * cosmo + 0.3f * comp;
        out[1] = matter;
        out[2] = cosmo;
        out[3] = comp;
    }
}

extern "C" void kernel_launch(void* const* d_in, const int* in_sizes, int n_in,
                              void* d_out, int out_size, void* d_ws, size_t ws_size,
                              hipStream_t stream) {
    const float* z1  = (const float*)d_in[0];
    const float* z2  = (const float*)d_in[1];
    const int*   mt1 = (const int*)d_in[2];
    const int*   mt2 = (const int*)d_in[3];
    // inputs 4..7 (cosmology/component) are unused by the reference math

    char* ws = (char*)d_ws;
    _Float16* aFrag  = (_Float16*)(ws);                                // 2 MB
    _Float16* bFrag  = (_Float16*)(ws + (size_t)(2u << 20));           // 4 MB
    float2*   maskf  = (float2*)  (ws + (size_t)(6u << 20));           // 64 KB
    float*    posbuf = (float*)   (ws + (size_t)(6u << 20) + (128u << 10)); // 16 KB
    float*    partial= (float*)   (ws + (size_t)(6u << 20) + (256u << 10)); // 1.5 MB

    prep_norm_kernel<<<TWO_B / 8, 256, 0, stream>>>(z1, z2, aFrag, bFrag);
    mask_kernel<<<B_SZ / 256, 256, 0, stream>>>(mt1, mt2, maskf);
    gemm_lse_kernel<<<NRPAIR * NCHUNK, 32, 0, stream>>>(aFrag, bFrag, maskf,
                                                        partial, posbuf);
    final_kernel<<<1, 256, 0, stream>>>(partial, posbuf, maskf, (float*)d_out);
}
